// CTRNN_50723563766278
// MI455X (gfx1250) — compile-verified, matrix operand order checked by reading, not run on hardware
//
#include <hip/hip_runtime.h>

// CTRNN scan for MI455X (gfx1250, wave32, WMMA bf16) — v2.
// SEQ=2048, BATCH=128, IN=128, HID=256, ALPHA=0.1
//
// 8 persistent workgroups, one per 16-row batch tile; full 2048-step
// recurrence with one __syncthreads per step. v2 change: all 24 weight
// B-fragments per wave are held in VGPRs (192 VGPRs/lane), loaded once from
// global at init. This removes 2/3 of the per-step LDS traffic (the loop was
// LDS-read bound: 300 KB/step -> ~108 KB/step) and eliminates the 192 KB LDS
// weight array. h state stays f32 in registers (D layout); a bf16 copy is
// exchanged through double-buffered LDS (A layout) each step; x tiles are
// staged one step ahead.

#define S_LEN   2048
#define BATCH_N 128
#define IN_D    128
#define HID_N   256
#define ALPHA_F 0.1f

typedef __attribute__((ext_vector_type(16))) __bf16       v16bf;
typedef __attribute__((ext_vector_type(8)))  float        v8f;
typedef __attribute__((ext_vector_type(8)))  unsigned int v8u;

__device__ __forceinline__ unsigned int f2bf_bits(float f) {
  unsigned int u = __float_as_uint(f);
  return (u + 0x7FFFu + ((u >> 16) & 1u)) >> 16;   // round-to-nearest-even
}
__device__ __forceinline__ unsigned int pack2bf(float lo, float hi) {
  return f2bf_bits(lo) | (f2bf_bits(hi) << 16);
}

__global__ __launch_bounds__(256)
void ctrnn_scan_wmma_bf16(const float* __restrict__ x,
                          const float* __restrict__ h0,
                          const float* __restrict__ W_in,
                          const float* __restrict__ b_in,
                          const float* __restrict__ W_h,
                          const float* __restrict__ b_h,
                          float* __restrict__ out) {
  __shared__ unsigned int lds_h[2][8 * 256];    // 16 KB  bf16 A-frags of h
  __shared__ unsigned int lds_x[2][4 * 256];    //  8 KB  bf16 A-frags of x tile
  __shared__ float        lds_bias[HID_N];      //  1 KB  b_in + b_h

  const int tid    = threadIdx.x;
  const int wg     = blockIdx.x;        // 0..7 : batch tile
  const int wgBase = wg * 16;
  const int wave   = tid >> 5;          // 0..7
  const int lane   = tid & 31;
  const int ln     = lane & 15;
  const int lh     = lane >> 4;

  // ---------------- init: fused bias ----------------
  if (tid < HID_N) lds_bias[tid] = b_in[tid] + b_h[tid];

  // ---------------- init: weight B-fragments -> registers ----------------
  // B-frag map (mirrors 16-bit A layout): lane holds column n = nt*16+ln;
  // vgpr v: g=v/4, w=v%4 -> k = kbase + g*16 + lh*8 + w*2 (+0,+1 packed).
  // ks 0..3 -> W_in (K=128), ks 4..11 -> W_h (K=256).
  v16bf Breg[2][12];
#pragma unroll
  for (int j = 0; j < 2; ++j) {
    int nn = (wave * 2 + j) * 16 + ln;
#pragma unroll
    for (int ks = 0; ks < 12; ++ks) {
      const float* base = (ks < 4) ? (W_in + nn * IN_D + ks * 32)
                                   : (W_h + nn * HID_N + (ks - 4) * 32);
      v8u u;
#pragma unroll
      for (int v = 0; v < 8; ++v) {
        int klocal = (v >> 2) * 16 + lh * 8 + (v & 3) * 2;
        u[v] = pack2bf(base[klocal], base[klocal + 1]);
      }
      Breg[j][ks] = __builtin_bit_cast(v16bf, u);
    }
  }

  // ---------------- init: h0 -> bf16 A-fragments (buffer 0) ----------------
  for (int i = tid; i < 8 * 256; i += 256) {
    int kt = i >> 8, within = i & 255;
    int laneA = within >> 3, v = within & 7;
    int M = laneA & 15, half = laneA >> 4, g = v >> 2, w = v & 3;
    int k = kt * 32 + g * 16 + half * 8 + w * 2;
    const float* p = h0 + (wgBase + M) * HID_N + k;
    lds_h[0][i] = pack2bf(p[0], p[1]);
  }

  // ---------------- init: x tile for t=0 -> lds_x[0] ----------------
  {
    int e = tid * 8, M = e >> 7, K = e & 127;   // 16x128 tile, 8 elems/thread
    const float* p = x + (size_t)(wgBase + M) * IN_D + K;
    float4 a = *(const float4*)p;
    float4 b = *(const float4*)(p + 4);
    uint4 u;
    u.x = pack2bf(a.x, a.y); u.y = pack2bf(a.z, a.w);
    u.z = pack2bf(b.x, b.y); u.w = pack2bf(b.z, b.w);
    int kt = K >> 5, klocal = K & 31, g = klocal >> 4, half = (klocal & 15) >> 3;
    int laneA = half * 16 + M;
    *(uint4*)&lds_x[0][kt * 256 + laneA * 8 + g * 4] = u;
  }

  // ---------------- init: f32 h state in registers (D layout) ----------
  float hA[8], hB[8];
  const int K0 = (wave * 2 + 0) * 16 + ln;
  const int K1 = (wave * 2 + 1) * 16 + ln;
#pragma unroll
  for (int vd = 0; vd < 8; ++vd) {
    int rowg = wgBase + vd + 8 * lh;
    hA[vd] = h0[rowg * HID_N + K0];
    hB[vd] = h0[rowg * HID_N + K1];
  }

  // A-layout ushort index bases for writing h bf16 at (row M, col K{0,1}).
  int idxBase0, idxBase1;
  {
    int kt = K0 >> 5, kl = K0 & 31, g = kl >> 4, r = kl & 15;
    int half = r >> 3, w = (r & 7) >> 1, lo = r & 1;
    idxBase0 = ((kt * 256 + half * 16 * 8 + g * 4 + w) << 1) + lo;
  }
  {
    int kt = K1 >> 5, kl = K1 & 31, g = kl >> 4, r = kl & 15;
    int half = r >> 3, w = (r & 7) >> 1, lo = r & 1;
    idxBase1 = ((kt * 256 + half * 16 * 8 + g * 4 + w) << 1) + lo;
  }

  __syncthreads();
  const float biasA = lds_bias[K0];
  const float biasB = lds_bias[K1];

  // ============================ main scan ============================
  for (int t = 0; t < S_LEN; ++t) {
    __syncthreads();                    // buffers (t&1) fully written

    // stage x tile for step t+1 into the other buffer (no dependency on h)
    if (t + 1 < S_LEN) {
      int e = tid * 8, M = e >> 7, K = e & 127;
      const float* p = x + ((size_t)(t + 1) * BATCH_N + wgBase + M) * IN_D + K;
      float4 a = *(const float4*)p;
      float4 b = *(const float4*)(p + 4);
      uint4 u;
      u.x = pack2bf(a.x, a.y); u.y = pack2bf(a.z, a.w);
      u.z = pack2bf(b.x, b.y); u.w = pack2bf(b.z, b.w);
      int kt = K >> 5, klocal = K & 31, g = klocal >> 4, half = (klocal & 15) >> 3;
      int laneA = half * 16 + M;
      *(uint4*)&lds_x[(t + 1) & 1][kt * 256 + laneA * 8 + g * 4] = u;
    }

    const unsigned int* hb = lds_h[t & 1];
    const unsigned int* xb = lds_x[t & 1];

    v8f c0 = {0.f, 0.f, 0.f, 0.f, 0.f, 0.f, 0.f, 0.f};
    v8f c1 = {0.f, 0.f, 0.f, 0.f, 0.f, 0.f, 0.f, 0.f};

#pragma unroll
    for (int ks = 0; ks < 12; ++ks) {
      v8u araw = (ks < 4) ? *(const v8u*)&xb[ks * 256 + lane * 8]
                          : *(const v8u*)&hb[(ks - 4) * 256 + lane * 8];
      v16bf A = __builtin_bit_cast(v16bf, araw);
      c0 = __builtin_amdgcn_wmma_f32_16x16x32_bf16(false, A, false, Breg[0][ks],
                                                   (short)0, c0, false, false);
      c1 = __builtin_amdgcn_wmma_f32_16x16x32_bf16(false, A, false, Breg[1][ks],
                                                   (short)0, c1, false, false);
    }

    // epilogue: bias + relu + leaky update, store outputs, refresh bf16 h
    unsigned short* hp = (unsigned short*)&lds_h[(t + 1) & 1][0];
    float* orow = out + (size_t)t * (BATCH_N * HID_N);
#pragma unroll
    for (int vd = 0; vd < 8; ++vd) {
      int M = vd + 8 * lh;
      int rowg = wgBase + M;
      float hn0 = fmaxf(c0[vd] + biasA, 0.0f);
      float hn1 = fmaxf(c1[vd] + biasB, 0.0f);
      hA[vd] = hA[vd] * (1.0f - ALPHA_F) + ALPHA_F * hn0;
      hB[vd] = hB[vd] * (1.0f - ALPHA_F) + ALPHA_F * hn1;
      orow[rowg * HID_N + K0] = hA[vd];
      orow[rowg * HID_N + K1] = hB[vd];
      hp[idxBase0 + M * 16] = (unsigned short)f2bf_bits(hA[vd]);
      hp[idxBase1 + M * 16] = (unsigned short)f2bf_bits(hB[vd]);
    }
  }

  // ---------------- h_final ----------------
  float* hf = out + (size_t)S_LEN * BATCH_N * HID_N;
#pragma unroll
  for (int vd = 0; vd < 8; ++vd) {
    int rowg = wgBase + vd + 8 * lh;
    hf[rowg * HID_N + K0] = hA[vd];
    hf[rowg * HID_N + K1] = hB[vd];
  }
}

extern "C" void kernel_launch(void* const* d_in, const int* in_sizes, int n_in,
                              void* d_out, int out_size, void* d_ws, size_t ws_size,
                              hipStream_t stream) {
  (void)in_sizes; (void)n_in; (void)out_size; (void)d_ws; (void)ws_size;
  const float* x    = (const float*)d_in[0];
  const float* h0   = (const float*)d_in[1];
  const float* W_in = (const float*)d_in[2];
  const float* b_in = (const float*)d_in[3];
  const float* W_h  = (const float*)d_in[4];
  const float* b_h  = (const float*)d_in[5];
  float* out = (float*)d_out;
  ctrnn_scan_wmma_bf16<<<BATCH_N / 16, 256, 0, stream>>>(x, h0, W_in, b_in,
                                                         W_h, b_h, out);
}
